// CantorAttention_26104811225652
// MI455X (gfx1250) — matrix operand order repaired and checked by reading, not compiled, not run on hardware
//
#include <hip/hip_runtime.h>

#define DIMW   512
#define NHEADS 8
#define HDIM   64
#define SEQ    2048
#define BATCH  2
#define KR     64
#define MROWS  (BATCH*SEQ)   // 4096

typedef __attribute__((ext_vector_type(16))) __bf16 v16bf;
typedef __attribute__((ext_vector_type(8)))  float  v8f;
typedef __attribute__((ext_vector_type(4)))  float  f4;

union FragBF { v16bf v; f4 f[2]; };
union BF8    { f4 f; unsigned short u[8]; };

__device__ __forceinline__ unsigned short f2bf(float x) {
    union { float f; unsigned u; } c; c.f = x;
    unsigned r = c.u + 0x7FFFu + ((c.u >> 16) & 1u);
    return (unsigned short)(r >> 16);
}
__device__ __forceinline__ float bf2f(unsigned short h) {
    union { unsigned u; float f; } c; c.u = ((unsigned)h) << 16; return c.f;
}

// ---------------- prep kernels ----------------
__global__ void cvt_bf16(const float* __restrict__ src, unsigned short* __restrict__ dst, int n) {
    int i = blockIdx.x * blockDim.x + threadIdx.x;
    if (i < n) dst[i] = f2bf(src[i]);
}

// w: K x N row-major (fp32)  ->  wt: N x K row-major (bf16)
__global__ void transpose_w(const float* __restrict__ w, unsigned short* __restrict__ wt, int K, int N) {
    int i = blockIdx.x * blockDim.x + threadIdx.x;
    if (i < K * N) {
        int n = i / K, k = i % K;
        wt[i] = f2bf(w[k * N + n]);
    }
}

// ---------------- WMMA GEMM core ----------------
// C(M x N) = A(M x K=512, bf16 row-major) * Bt(N x K, bf16 row-major)^T + bias
// MODE 0: scatter into Q/K/V bf16 [B,H,S,D]   (N_DIM = 1536)
// MODE 1: store fp32 to fo (row*N_DIM + col)  (N_DIM = 512)
// K-loop is explicitly double-buffered in registers so each v_wmma consumes
// fragments whose loads were issued one K-step earlier (L2-latency hiding
// within a wave, not just across waves).
template <int N_DIM, int MODE>
__global__ __launch_bounds__(256)
void gemm_wmma(const unsigned short* __restrict__ A,
               const unsigned short* __restrict__ Bt,
               const float* __restrict__ bias,
               unsigned short* __restrict__ qo,
               unsigned short* __restrict__ ko,
               unsigned short* __restrict__ vo,
               float* __restrict__ fo)
{
    const int K = DIMW;
    const int lane = threadIdx.x & 31;
    const int wave = threadIdx.x >> 5;
    const int wm = wave & 3;           // 4 waves along M
    const int wn = wave >> 2;          // 2 waves along N
    const int m0 = blockIdx.y * 64 + wm * 16;
    const int n0 = blockIdx.x * 128 + wn * 64;
    const int half = lane >> 4;
    const int l15  = lane & 15;

    v8f acc[4];
#pragma unroll
    for (int t = 0; t < 4; ++t)
#pragma unroll
        for (int i = 0; i < 8; ++i) acc[t][i] = 0.0f;

    const unsigned short* aRow = A + (size_t)(m0 + l15) * K;
    const unsigned short* bRow[4];
#pragma unroll
    for (int t = 0; t < 4; ++t)
        bRow[t] = Bt + (size_t)(n0 + t * 16 + l15) * K;

    // ---- prologue: load K-step 0 fragments ----
    FragBF aC, bC[4];
    // 16-bit A 16x32 layout: VGPR0-3 = K[half*8 .. +7], VGPR4-7 = K[16+half*8 ..]
    aC.f[0] = *(const f4*)(aRow + half * 8);
    aC.f[1] = *(const f4*)(aRow + 16 + half * 8);
#pragma unroll
    for (int t = 0; t < 4; ++t) {
        // 16-bit B 32x16 layout: VGPR0-3 = K[half*8 .. +7], VGPR4-7 = K[16+half*8 ..]
        bC[t].f[0] = *(const f4*)(bRow[t] + half * 8);
        bC[t].f[1] = *(const f4*)(bRow[t] + 16 + half * 8);
    }

#pragma unroll
    for (int kt = 0; kt < K; kt += 32) {
        FragBF aN, bN[4];
        const int kn = kt + 32;
        if (kn < K) {                       // compile-time folded (full unroll)
            aN.f[0] = *(const f4*)(aRow + kn + half * 8);
            aN.f[1] = *(const f4*)(aRow + kn + 16 + half * 8);
#pragma unroll
            for (int t = 0; t < 4; ++t) {
                bN[t].f[0] = *(const f4*)(bRow[t] + kn + half * 8);
                bN[t].f[1] = *(const f4*)(bRow[t] + kn + 16 + half * 8);
            }
        }
#pragma unroll
        for (int t = 0; t < 4; ++t)
            acc[t] = __builtin_amdgcn_wmma_f32_16x16x32_bf16(
                false, aC.v, false, bC[t].v, (short)0, acc[t], false, false);
        if (kn < K) {
            aC = aN;
#pragma unroll
            for (int t = 0; t < 4; ++t) bC[t] = bN[t];
        }
    }

#pragma unroll
    for (int t = 0; t < 4; ++t) {
        const int col = n0 + t * 16 + l15;
        const float bv = bias[col];
#pragma unroll
        for (int r = 0; r < 8; ++r) {
            const int row = m0 + r + 8 * half;     // C layout: VGPR r, half selects M / M+8
            const float val = acc[t][r] + bv;
            if (MODE == 0) {
                const int b = row >> 11, s = row & (SEQ - 1);
                const int sel = col >> 9, w = col & 511;
                const int h = w >> 6, d = w & 63;
                unsigned short* dst = (sel == 0) ? qo : (sel == 1) ? ko : vo;
                dst[(((size_t)b * NHEADS + h) * SEQ + s) * HDIM + d] = f2bf(val);
            } else {
                fo[(size_t)row * N_DIM + col] = val;
            }
        }
    }
}

// ---------------- gather attention ----------------
// one wave per (b,h,q); lanes over routes for scores, lanes over d for PV.
__global__ __launch_bounds__(256)
void attn_kernel(const unsigned short* __restrict__ Qb,
                 const unsigned short* __restrict__ Kb,
                 const unsigned short* __restrict__ Vb,
                 const int* __restrict__ routes,
                 unsigned short* __restrict__ aout)
{
    __shared__ float qs[8][HDIM];
    __shared__ float at[8][KR];
    __shared__ int   rt[8][KR];

    const int lane = threadIdx.x & 31;
    const int wv   = threadIdx.x >> 5;
    const int gid  = blockIdx.x * 8 + wv;      // gid = bh*SEQ + q
    const int q    = gid & (SEQ - 1);
    const int bh   = gid >> 11;

    const unsigned short* kBase = Kb + (size_t)bh * SEQ * HDIM;
    const unsigned short* vBase = Vb + (size_t)bh * SEQ * HDIM;

    { // stage q-vector (fp32) into LDS
        const unsigned short* qp = Qb + ((size_t)bh * SEQ + q) * HDIM;
        qs[wv][lane]      = bf2f(qp[lane]);
        qs[wv][lane + 32] = bf2f(qp[lane + 32]);
    }
    const int r0 = routes[q * KR + lane];
    const int r1 = routes[q * KR + lane + 32];
    rt[wv][lane] = r0; rt[wv][lane + 32] = r1;
    __syncthreads();

    // scores: each lane handles 2 routes, 16B gathers of K
    float s0 = 0.f, s1 = 0.f;
    const unsigned short* kp0 = kBase + (size_t)r0 * HDIM;
    const unsigned short* kp1 = kBase + (size_t)r1 * HDIM;
#pragma unroll
    for (int d = 0; d < HDIM; d += 8) {
        BF8 k0, k1;
        k0.f = *(const f4*)(kp0 + d);
        k1.f = *(const f4*)(kp1 + d);
#pragma unroll
        for (int e = 0; e < 8; ++e) {
            const float qv = qs[wv][d + e];
            s0 = fmaf(qv, bf2f(k0.u[e]), s0);
            s1 = fmaf(qv, bf2f(k1.u[e]), s1);
        }
    }
    const float scale = 0.125f;               // 1/sqrt(64)
    s0 *= scale; s1 *= scale;

    float mx = fmaxf(s0, s1);
#pragma unroll
    for (int o = 16; o > 0; o >>= 1) mx = fmaxf(mx, __shfl_xor(mx, o, 32));
    const float e0 = __expf(s0 - mx), e1 = __expf(s1 - mx);
    float sum = e0 + e1;
#pragma unroll
    for (int o = 16; o > 0; o >>= 1) sum += __shfl_xor(sum, o, 32);
    const float inv = 1.f / sum;
    at[wv][lane] = e0 * inv; at[wv][lane + 32] = e1 * inv;
    __syncthreads();

    // out: lanes over d (2 contiguous elems per lane => coalesced V reads)
    float o0 = 0.f, o1 = 0.f;
    const int d0 = lane * 2;
#pragma unroll 8
    for (int j = 0; j < KR; ++j) {
        const float a = at[wv][j];
        const int rj = rt[wv][j];
        const unsigned vv = *(const unsigned*)(vBase + (size_t)rj * HDIM + d0);
        o0 = fmaf(a, bf2f((unsigned short)(vv & 0xFFFFu)), o0);
        o1 = fmaf(a, bf2f((unsigned short)(vv >> 16)), o1);
    }
    const int b = bh >> 3, h = bh & 7;
    const size_t row = (size_t)b * SEQ + q;
    const unsigned packed = (unsigned)f2bf(o0) | ((unsigned)f2bf(o1) << 16);
    *(unsigned*)(aout + row * DIMW + h * HDIM + d0) = packed;
}

// ---------------- launcher ----------------
extern "C" void kernel_launch(void* const* d_in, const int* in_sizes, int n_in,
                              void* d_out, int out_size, void* d_ws, size_t ws_size,
                              hipStream_t stream)
{
    const float* x     = (const float*)d_in[0];
    const int*   routes= (const int*)  d_in[1];
    const float* w_qkv = (const float*)d_in[2];
    const float* b_qkv = (const float*)d_in[3];
    const float* w_out = (const float*)d_in[4];
    const float* b_out = (const float*)d_in[5];
    float* out = (float*)d_out;

    char* ws = (char*)d_ws;
    size_t off = 0;
    auto alloc = [&](size_t bytes) -> char* {
        char* p = ws + off;
        off += (bytes + 255) & ~(size_t)255;
        return p;
    };
    unsigned short* xbf = (unsigned short*)alloc((size_t)MROWS * DIMW * 2);
    unsigned short* wtq = (unsigned short*)alloc((size_t)3 * DIMW * DIMW * 2);
    unsigned short* wto = (unsigned short*)alloc((size_t)DIMW * DIMW * 2);
    unsigned short* Qb  = (unsigned short*)alloc((size_t)BATCH * NHEADS * SEQ * HDIM * 2);
    unsigned short* Kb  = (unsigned short*)alloc((size_t)BATCH * NHEADS * SEQ * HDIM * 2);
    unsigned short* Vb  = (unsigned short*)alloc((size_t)BATCH * NHEADS * SEQ * HDIM * 2);
    unsigned short* ao  = (unsigned short*)alloc((size_t)MROWS * DIMW * 2);

    const int nx = MROWS * DIMW;
    cvt_bf16<<<(nx + 255) / 256, 256, 0, stream>>>(x, xbf, nx);
    transpose_w<<<(3 * DIMW * DIMW + 255) / 256, 256, 0, stream>>>(w_qkv, wtq, DIMW, 3 * DIMW);
    transpose_w<<<(DIMW * DIMW + 255) / 256, 256, 0, stream>>>(w_out, wto, DIMW, DIMW);

    dim3 g1(1536 / 128, MROWS / 64);  // 12 x 64 workgroups
    gemm_wmma<1536, 0><<<g1, 256, 0, stream>>>(xbf, wtq, b_qkv, Qb, Kb, Vb, nullptr);

    attn_kernel<<<(BATCH * NHEADS * SEQ) / 8, 256, 0, stream>>>(Qb, Kb, Vb, routes, ao);

    dim3 g2(512 / 128, MROWS / 64);   // 4 x 64 workgroups
    gemm_wmma<512, 1><<<g2, 256, 0, stream>>>(ao, wto, b_out, nullptr, nullptr, nullptr, out);
}
